// FrequencyAwareRouter_18442589569333
// MI455X (gfx1250) — compile-verified, matrix operand order checked by reading, not run on hardware
//
#include <hip/hip_runtime.h>
#include <hip/hip_bf16.h>
#include <math.h>

// ---------------------------------------------------------------------------
// FrequencyAwareRouter on MI455X (gfx1250)
//   FFT-as-GEMM with bf16 hi/lo split ("bf16x3") on v_wmma_f32_16x16x32_bf16.
//   One workgroup = one (image, 32-column v-strip); each wave job computes
//   two 16-wide v-subtiles so every A-fragment (x row / twiddle row) feeds
//   2x the output columns -> halves L2 traffic per WMMA vs a 16-wide strip.
//   Stage A: Y = x @ W256 (column DFT) -> bf16 hi/lo strip in LDS (transposed)
//   Stage B: F = W256 @ Y (complex)    -> magnitudes -> radial bin sums
//   Tiny router kernel: MLP + softmax + top-2 + re-softmax.
// ---------------------------------------------------------------------------

typedef __attribute__((ext_vector_type(16))) __bf16 v16bf;
typedef __attribute__((ext_vector_type(8)))  __bf16 bhalf8;   // 16 B
typedef __attribute__((ext_vector_type(8)))  float  v8f;

#define NSIDE  256
#define NPIX   65536
#define BINS   8
#define STRIPW 32
#define YSTR   264   // LDS row stride in bf16: 528 B % 256 B = 16 -> bank-spread, 16B aligned
#define YARR   (STRIPW * YSTR)                      // elements per Y array
#define SMEM_BYTES (4 * YARR * 2 + 32)              // 4 bf16 arrays + binAcc

struct BF16x16 { uint4 a, b; };

__device__ __forceinline__ v8f wmma_bf16(v16bf a, v16bf b, v8f c) {
    // (neg_a, A, neg_b, B, c_mod, C, reuse_a, reuse_b)
    return __builtin_amdgcn_wmma_f32_16x16x32_bf16(false, a, false, b, (short)0, c, false, false);
}

// bf16 fragment (global table row or LDS strip row):
// contiguous runs {base..base+7} and {base+16..base+23}
__device__ __forceinline__ v16bf load_bf16_frag(const __bf16* rowptr, int base) {
    BF16x16 r;
    r.a = *(const uint4*)(rowptr + base);
    r.b = *(const uint4*)(rowptr + base + 16);
    return __builtin_bit_cast(v16bf, r);
}

// f32 fragment (global image row) -> bf16 hi/lo fragments
__device__ __forceinline__ void load_f32_frag(const float* p, int base, v16bf& hi, v16bf& lo) {
    float4 f0 = *(const float4*)(p + base);
    float4 f1 = *(const float4*)(p + base + 4);
    float4 f2 = *(const float4*)(p + base + 16);
    float4 f3 = *(const float4*)(p + base + 20);
    float vals[16] = { f0.x, f0.y, f0.z, f0.w,  f1.x, f1.y, f1.z, f1.w,
                       f2.x, f2.y, f2.z, f2.w,  f3.x, f3.y, f3.z, f3.w };
#pragma unroll
    for (int i = 0; i < 16; ++i) {
        __bf16 h = (__bf16)vals[i];
        hi[i] = h;
        lo[i] = (__bf16)(vals[i] - (float)h);
    }
}

// ---------------------------------------------------------------------------
// Setup: cos/sin DFT tables (bf16 hi + lo residual), radial bin table,
// and zero the energy accumulator (must happen every launch).
// ---------------------------------------------------------------------------
__global__ void far_build_tables(__bf16* __restrict__ cth, __bf16* __restrict__ ctl,
                                 __bf16* __restrict__ sth, __bf16* __restrict__ stl,
                                 int* __restrict__ binTab, float* __restrict__ energy)
{
    int idx = blockIdx.x * blockDim.x + threadIdx.x;   // 0..65535
    if (idx < 16 * BINS) energy[idx] = 0.0f;

    int j = idx >> 8, k = idx & 255;
    int prod = (j * k) & 255;                          // e^{-2pi i jk/256} periodic
    float ang = (float)prod * (6.2831853071795864f / 256.0f);
    float s, c;
    __sincosf(ang, &s, &c);
    __bf16 ch = (__bf16)c;  cth[idx] = ch;  ctl[idx] = (__bf16)(c - (float)ch);
    __bf16 sh = (__bf16)s;  sth[idx] = sh;  stl[idx] = (__bf16)(s - (float)sh);

    // bin table: distance grid uses (i-128, j-128) in natural (unshifted) order
    float dy = (float)(j - 128), dx = (float)(k - 128);
    float d = sqrtf(dy * dy + dx * dx);
    float dmax = sqrtf(32768.0f);
    int bin;
    if (d >= dmax) bin = BINS;                         // d == max falls in no bin
    else {
        bin = (int)floorf(d * (8.0f / dmax));          // uniform edges, searchsorted 'right'
        if (bin > BINS) bin = BINS;
    }
    binTab[idx] = bin;
}

// ---------------------------------------------------------------------------
// Main kernel: one workgroup = one (image, 32-column v-strip).
// Dynamic LDS: 4 x (32 x YSTR) bf16 Y arrays (67.6 KB; CDNA5 WGs may use up
// to 320 KB LDS) + 8-float bin accumulator.
// ---------------------------------------------------------------------------
__global__ __launch_bounds__(256) void far_fft_bin_kernel(
    const float*  __restrict__ x,
    const __bf16* __restrict__ cth, const __bf16* __restrict__ ctl,
    const __bf16* __restrict__ sth, const __bf16* __restrict__ stl,
    const int*    __restrict__ binTab,
    float*        __restrict__ energy)
{
    extern __shared__ char smem[];
    __bf16* yrh = (__bf16*)smem;              // Y real hi, transposed [v][y]
    __bf16* yrl = yrh + YARR;                 // Y real lo
    __bf16* yih = yrl + YARR;                 // Y imag hi
    __bf16* yil = yih + YARR;                 // Y imag lo
    float*  binAcc = (float*)(smem + 4 * YARR * 2);

    const int tid   = threadIdx.x;
    const int lane  = tid & 31;
    const int wave  = tid >> 5;
    const int img   = blockIdx.x >> 3;        // 0..1023 = b*64 + c
    const int strip = blockIdx.x & 7;
    const int v0    = strip * STRIPW;
    const int bidx  = img >> 6;               // batch index

    if (tid < BINS) binAcc[tid] = 0.0f;

    const int Mrow  = lane & 15;              // A-row / B-col owned by this lane
    const int kbase = (lane >> 4) * 8;        // K sub-block base per ISA layout
    const float* ximg = x + (size_t)img * NPIX;

    // -------- Stage A: Y[:, v0:v0+32] = x @ W   (Yr = x@C, Yi = -x@S) --------
    for (int j = 0; j < 2; ++j) {
        const int m  = wave + j * 8;          // 16 row-tiles over 8 waves
        const int y0 = m * 16;
        v8f aR0 = {}, aI0 = {}, aR1 = {}, aI1 = {};
        const float*  arow = ximg + (y0 + Mrow) * NSIDE;
        // symmetric table: row index = output column
        const __bf16* bch0 = cth + (v0 + Mrow) * NSIDE;
        const __bf16* bcl0 = ctl + (v0 + Mrow) * NSIDE;
        const __bf16* bsh0 = sth + (v0 + Mrow) * NSIDE;
        const __bf16* bsl0 = stl + (v0 + Mrow) * NSIDE;
        const int o1 = 16 * NSIDE;            // second v-subtile rows
        for (int kb = 0; kb < 8; ++kb) {
            const int k0 = kb * 32;
            v16bf ahi, alo;
            load_f32_frag(arow + k0, kbase, ahi, alo);   // shared by both subtiles
            v16bf c0h = load_bf16_frag(bch0 + k0, kbase);
            v16bf c0l = load_bf16_frag(bcl0 + k0, kbase);
            v16bf s0h = load_bf16_frag(bsh0 + k0, kbase);
            v16bf s0l = load_bf16_frag(bsl0 + k0, kbase);
            v16bf c1h = load_bf16_frag(bch0 + o1 + k0, kbase);
            v16bf c1l = load_bf16_frag(bcl0 + o1 + k0, kbase);
            v16bf s1h = load_bf16_frag(bsh0 + o1 + k0, kbase);
            v16bf s1l = load_bf16_frag(bsl0 + o1 + k0, kbase);
            aR0 = wmma_bf16(ahi, c0h, aR0);
            aR0 = wmma_bf16(ahi, c0l, aR0);
            aR0 = wmma_bf16(alo, c0h, aR0);
            aI0 = wmma_bf16(ahi, s0h, aI0);
            aI0 = wmma_bf16(ahi, s0l, aI0);
            aI0 = wmma_bf16(alo, s0h, aI0);
            aR1 = wmma_bf16(ahi, c1h, aR1);
            aR1 = wmma_bf16(ahi, c1l, aR1);
            aR1 = wmma_bf16(alo, c1h, aR1);
            aI1 = wmma_bf16(ahi, s1h, aI1);
            aI1 = wmma_bf16(ahi, s1l, aI1);
            aI1 = wmma_bf16(alo, s1h, aI1);
        }
        // D element g lives at (M = g + 8*(lane>>4), N = lane&15).
        // Split to bf16 hi/lo once, store transposed: b128 DS stores.
        const int ybase = y0 + ((lane >> 4) * 8);
#pragma unroll
        for (int s = 0; s < 2; ++s) {
            const v8f& aR = s ? aR1 : aR0;
            const v8f& aI = s ? aI1 : aI0;
            bhalf8 rh, rl, ih, il;
#pragma unroll
            for (int g = 0; g < 8; ++g) {
                float vr = aR[g];
                __bf16 hr = (__bf16)vr;
                rh[g] = hr;  rl[g] = (__bf16)(vr - (float)hr);
                float vi = -aI[g];                       // Yi = -(x @ S)
                __bf16 hi_ = (__bf16)vi;
                ih[g] = hi_; il[g] = (__bf16)(vi - (float)hi_);
            }
            const int row = (s * 16 + Mrow) * YSTR + ybase;
            *(bhalf8*)&yrh[row] = rh;
            *(bhalf8*)&yrl[row] = rl;
            *(bhalf8*)&yih[row] = ih;
            *(bhalf8*)&yil[row] = il;
        }
    }
    __syncthreads();

    // -------- Stage B: F = W @ Y; magnitudes; radial bin accumulation --------
    float lacc[BINS];
#pragma unroll
    for (int b = 0; b < BINS; ++b) lacc[b] = 0.0f;

    for (int j = 0; j < 2; ++j) {
        const int t  = wave + j * 8;
        const int u0 = t * 16;
        // per v-subtile: Fr accumulator, C@Yi, S@Yr (Fi = p1 - p2)
        v8f fR0 = {}, p10 = {}, p20 = {};
        v8f fR1 = {}, p11 = {}, p21 = {};
        const __bf16* ach = cth + (u0 + Mrow) * NSIDE;
        const __bf16* acl = ctl + (u0 + Mrow) * NSIDE;
        const __bf16* ash = sth + (u0 + Mrow) * NSIDE;
        const __bf16* asl = stl + (u0 + Mrow) * NSIDE;
        const int r0 = Mrow * YSTR;
        const int r1 = (16 + Mrow) * YSTR;
        for (int kb = 0; kb < 8; ++kb) {
            const int k0 = kb * 32;
            v16bf achi = load_bf16_frag(ach + k0, kbase);   // shared by both subtiles
            v16bf aclo = load_bf16_frag(acl + k0, kbase);
            v16bf ashi = load_bf16_frag(ash + k0, kbase);
            v16bf aslo = load_bf16_frag(asl + k0, kbase);
            v16bf yr0h = load_bf16_frag(yrh + r0 + k0, kbase);
            v16bf yr0l = load_bf16_frag(yrl + r0 + k0, kbase);
            v16bf yi0h = load_bf16_frag(yih + r0 + k0, kbase);
            v16bf yi0l = load_bf16_frag(yil + r0 + k0, kbase);
            v16bf yr1h = load_bf16_frag(yrh + r1 + k0, kbase);
            v16bf yr1l = load_bf16_frag(yrl + r1 + k0, kbase);
            v16bf yi1h = load_bf16_frag(yih + r1 + k0, kbase);
            v16bf yi1l = load_bf16_frag(yil + r1 + k0, kbase);
            // subtile 0: Fr = C@Yr + S@Yi ; Fi = C@Yi - S@Yr
            fR0 = wmma_bf16(achi, yr0h, fR0);
            fR0 = wmma_bf16(achi, yr0l, fR0);
            fR0 = wmma_bf16(aclo, yr0h, fR0);
            fR0 = wmma_bf16(ashi, yi0h, fR0);
            fR0 = wmma_bf16(ashi, yi0l, fR0);
            fR0 = wmma_bf16(aslo, yi0h, fR0);
            p10 = wmma_bf16(achi, yi0h, p10);
            p10 = wmma_bf16(achi, yi0l, p10);
            p10 = wmma_bf16(aclo, yi0h, p10);
            p20 = wmma_bf16(ashi, yr0h, p20);
            p20 = wmma_bf16(ashi, yr0l, p20);
            p20 = wmma_bf16(aslo, yr0h, p20);
            // subtile 1
            fR1 = wmma_bf16(achi, yr1h, fR1);
            fR1 = wmma_bf16(achi, yr1l, fR1);
            fR1 = wmma_bf16(aclo, yr1h, fR1);
            fR1 = wmma_bf16(ashi, yi1h, fR1);
            fR1 = wmma_bf16(ashi, yi1l, fR1);
            fR1 = wmma_bf16(aslo, yi1h, fR1);
            p11 = wmma_bf16(achi, yi1h, p11);
            p11 = wmma_bf16(achi, yi1l, p11);
            p11 = wmma_bf16(aclo, yi1h, p11);
            p21 = wmma_bf16(ashi, yr1h, p21);
            p21 = wmma_bf16(ashi, yr1l, p21);
            p21 = wmma_bf16(aslo, yr1h, p21);
        }
        const int ur = u0 + ((lane >> 4) * 8);
#pragma unroll
        for (int s = 0; s < 2; ++s) {
            const v8f& fR = s ? fR1 : fR0;
            const v8f& p1 = s ? p11 : p10;
            const v8f& p2 = s ? p21 : p20;
            const int vcol = v0 + s * 16 + Mrow;
#pragma unroll
            for (int g = 0; g < 8; ++g) {
                float fr  = fR[g];
                float fi  = p1[g] - p2[g];
                float mag = sqrtf(fr * fr + fi * fi) * (1.0f / 65536.0f); // norm='forward'
                int bin   = binTab[(ur + g) * NSIDE + vcol];
#pragma unroll
                for (int b = 0; b < BINS; ++b)
                    lacc[b] += (bin == b) ? mag : 0.0f;
            }
        }
    }

    // wave reduce -> LDS -> one global atomic per (batch, bin)
#pragma unroll
    for (int b = 0; b < BINS; ++b) {
        float v = lacc[b];
        for (int off = 16; off > 0; off >>= 1) v += __shfl_xor(v, off, 32);
        if (lane == 0) atomicAdd(&binAcc[b], v);
    }
    __syncthreads();
    if (tid < BINS)
        atomicAdd(&energy[bidx * BINS + tid], binAcc[tid] * (1.0f / 64.0f)); // channel mean
}

// ---------------------------------------------------------------------------
// Router: MLP + softmax + top-2 + re-softmax. One lane per batch row.
// ---------------------------------------------------------------------------
__global__ void far_router_kernel(const float* __restrict__ energy,
                                  const float* __restrict__ W1, const float* __restrict__ b1,
                                  const float* __restrict__ W2, const float* __restrict__ b2,
                                  float* __restrict__ out)
{
    int b = threadIdx.x;
    if (b >= 16) return;
    float e[8];
#pragma unroll
    for (int k = 0; k < 8; ++k) e[k] = energy[b * 8 + k];
    float h[32];
#pragma unroll
    for (int j = 0; j < 32; ++j) {
        float s = b1[j];
#pragma unroll
        for (int k = 0; k < 8; ++k) s += W1[j * 8 + k] * e[k];
        h[j] = fmaxf(s, 0.0f);
    }
    float l[4];
#pragma unroll
    for (int o = 0; o < 4; ++o) {
        float s = b2[o];
#pragma unroll
        for (int j = 0; j < 32; ++j) s += W2[o * 32 + j] * h[j];
        l[o] = s;
    }
    float mx = fmaxf(fmaxf(l[0], l[1]), fmaxf(l[2], l[3]));
    float p[4], sum = 0.0f;
#pragma unroll
    for (int o = 0; o < 4; ++o) { p[o] = expf(l[o] - mx); sum += p[o]; }
#pragma unroll
    for (int o = 0; o < 4; ++o) p[o] /= sum;
    int i0 = 0;
#pragma unroll
    for (int o = 1; o < 4; ++o) if (p[o] > p[i0]) i0 = o;          // ties -> lower idx
    int i1 = (i0 == 0) ? 1 : 0;
#pragma unroll
    for (int o = 0; o < 4; ++o) if (o != i0 && p[o] > p[i1]) i1 = o;
    float z  = expf(p[i1] - p[i0]);                                 // softmax over top-2
    float s2 = 1.0f + z;
    out[b * 2 + 0]      = (float)i0;
    out[b * 2 + 1]      = (float)i1;
    out[32 + b * 2 + 0] = 1.0f / s2;
    out[32 + b * 2 + 1] = z / s2;
}

// ---------------------------------------------------------------------------
extern "C" void kernel_launch(void* const* d_in, const int* in_sizes, int n_in,
                              void* d_out, int out_size, void* d_ws, size_t ws_size,
                              hipStream_t stream)
{
    (void)in_sizes; (void)n_in; (void)out_size; (void)ws_size;
    const float* x  = (const float*)d_in[0];
    const float* W1 = (const float*)d_in[1];
    const float* b1 = (const float*)d_in[2];
    const float* W2 = (const float*)d_in[3];
    const float* b2 = (const float*)d_in[4];
    // d_in[5] = k (always 2)

    char* ws = (char*)d_ws;
    float*  energy = (float*)ws;                         // 128 f32
    int*    binTab = (int*)(ws + 512);                   // 65536 i32
    __bf16* cth    = (__bf16*)(ws + 512 + 262144);       // 4 x 65536 bf16
    __bf16* ctl    = cth + NPIX;
    __bf16* sth    = ctl + NPIX;
    __bf16* stl    = sth + NPIX;

    far_build_tables<<<256, 256, 0, stream>>>(cth, ctl, sth, stl, binTab, energy);
    far_fft_bin_kernel<<<1024 * 8, 256, SMEM_BYTES, stream>>>(x, cth, ctl, sth, stl, binTab, energy);
    far_router_kernel<<<1, 32, 0, stream>>>(energy, W1, b1, W2, b2, (float*)d_out);
}
